// Head_28269474742902
// MI455X (gfx1250) — compile-verified, matrix operand order checked by reading, not run on hardware
//
#include <hip/hip_runtime.h>

// Shapes fixed by the reference: B=32, T=1024, C=1024, HD=64.
#define B_  32
#define T_  1024
#define C_  1024
#define HD_ 64
// Workspace: q, k, v as bf16 [B*T][HD] -> 3 * 32768*64*2 = 12 MB.

typedef __attribute__((ext_vector_type(16))) __bf16        v16bf;
typedef __attribute__((ext_vector_type(2)))  __bf16        v2bf;
typedef __attribute__((ext_vector_type(8)))  float         v8f;
typedef __attribute__((ext_vector_type(8)))  unsigned int  v8u;

// f32 -> bf16 via native fptrunc: the backend folds a 2-wide build-vector of
// these into v_cvt_pk_bf16_f32 when the target has it (RNE either way).
__device__ __forceinline__ unsigned int pk2(float a, float b) {
  v2bf r;
  r[0] = (__bf16)a;
  r[1] = (__bf16)b;
  return __builtin_bit_cast(unsigned int, r);
}
__device__ __forceinline__ unsigned short f2bf(float f) {
  return __builtin_bit_cast(unsigned short, (__bf16)f);
}

// ---- CDNA5 async global->LDS copy (ASYNCcnt-tracked), via inline asm --------
// VDST = per-lane LDS byte address (low 32 bits of a generic pointer to LDS),
// VADDR = 64-bit global address, saddr = off.
#define ASYNC_CP_B128(dst_lds, src_glb)                                        \
  asm volatile("global_load_async_to_lds_b128 %0, %1, off"                     \
               :: "v"((unsigned int)(unsigned long long)(const void*)(dst_lds)),\
                  "v"((const void*)(src_glb))                                  \
               : "memory")
#define WAIT_ASYNC0() asm volatile("s_wait_asynccnt 0" ::: "memory")
#define WAIT_DS0()    asm volatile("s_wait_dscnt 0" ::: "memory")

// ---------------------------------------------------------------------------
// Kernel 1: q/k/v projection. grid = (BT/128, 3), block = 256 (8 waves).
// Each wave computes a 16x64 tile via v_wmma_f32_16x16x32_bf16.
// ---------------------------------------------------------------------------
__global__ __launch_bounds__(256) void qkv_proj_kernel(
    const float* __restrict__ x,  const float* __restrict__ Wq,
    const float* __restrict__ Wk, const float* __restrict__ Wv,
    unsigned short* __restrict__ qw, unsigned short* __restrict__ kw,
    unsigned short* __restrict__ vw)
{
  __shared__ __align__(16) unsigned short xt[128 * 32];  // x tile, bf16 [row][k]
  __shared__ __align__(16) unsigned short wt[64 * 32];   // W tile, bf16 [n][k]

  const int tid   = threadIdx.x;
  const int wave  = tid >> 5;
  const int lane  = tid & 31;
  const int l16   = lane & 15;
  const int lhalf = lane >> 4;
  const int blk   = blockIdx.x;
  const int which = blockIdx.y;

  const float*    W   = (which == 0) ? Wq : (which == 1) ? Wk : Wv;
  unsigned short* dst = (which == 0) ? qw : (which == 1) ? kw : vw;

  // K-pair base offsets for the 16-bit 16x32 A/B fragment layout (ISA 7.12.2).
  int kofs[8];
#pragma unroll
  for (int j = 0; j < 8; ++j) kofs[j] = 2 * j + ((j >= 4) ? 8 : 0) + lhalf * 8;

  v8f acc[4];
#pragma unroll
  for (int t = 0; t < 4; ++t)
#pragma unroll
    for (int r = 0; r < 8; ++r) acc[t][r] = 0.f;

  const size_t rowbase = (size_t)blk * 128;

  for (int kb = 0; kb < C_ / 32; ++kb) {
    const int k0 = kb * 32;
    __syncthreads();
    // Stage x tile: 128 rows x 32 K (f32 -> bf16), 4 float4 per thread.
#pragma unroll
    for (int p = 0; p < 4; ++p) {
      int idx = tid + p * 256;
      int row = idx >> 3, c4 = idx & 7;
      const float4 v = *(const float4*)(x + (rowbase + row) * C_ + k0 + c4 * 4);
      unsigned int* q = (unsigned int*)&xt[row * 32 + c4 * 4];
      q[0] = pk2(v.x, v.y);
      q[1] = pk2(v.z, v.w);
    }
    // Stage W tile: 64 rows x 32 K, 2 float4 per thread.
#pragma unroll
    for (int p = 0; p < 2; ++p) {
      int idx = tid + p * 256;
      int n = idx >> 3, c4 = idx & 7;
      const float4 v = *(const float4*)(W + (size_t)n * C_ + k0 + c4 * 4);
      unsigned int* q = (unsigned int*)&wt[n * 32 + c4 * 4];
      q[0] = pk2(v.x, v.y);
      q[1] = pk2(v.z, v.w);
    }
    // Prefetch next x K-tile while this one is consumed.
    if (kb + 1 < C_ / 32)
      __builtin_prefetch(x + (rowbase + (tid >> 1)) * C_ + k0 + 32 + (tid & 1) * 16, 0, 0);
    __syncthreads();

    // A fragment: rows (wave*16 + M), M = l16.
    v8u af;
    const unsigned short* xr = &xt[(wave * 16 + l16) * 32];
#pragma unroll
    for (int j = 0; j < 8; ++j) af[j] = *(const unsigned int*)(xr + kofs[j]);
    const v16bf A = __builtin_bit_cast(v16bf, af);

#pragma unroll
    for (int t = 0; t < 4; ++t) {
      v8u bfv;
      const unsigned short* wr = &wt[(t * 16 + l16) * 32];
#pragma unroll
      for (int j = 0; j < 8; ++j) bfv[j] = *(const unsigned int*)(wr + kofs[j]);
      const v16bf Bm = __builtin_bit_cast(v16bf, bfv);
      acc[t] = __builtin_amdgcn_wmma_f32_16x16x32_bf16(
          false, A, false, Bm, (short)0, acc[t], false, false);
    }
  }

  // C/D layout: lane -> N = l16, VGPR r -> M = r + 8*lhalf.
#pragma unroll
  for (int t = 0; t < 4; ++t)
#pragma unroll
    for (int r = 0; r < 8; ++r) {
      size_t row = rowbase + wave * 16 + r + 8 * lhalf;
      dst[row * HD_ + t * 16 + l16] = f2bf(acc[t][r]);
    }
}

// ---------------------------------------------------------------------------
// Kernel 2: causal flash attention over bf16 q/k/v, f32 output.
// grid = (T/128, B), block = 256 (8 waves, 16 query rows each).
// Double-buffered async global->LDS staging of k/v blocks (32 keys each).
// ---------------------------------------------------------------------------
__global__ __launch_bounds__(256) void attn_kernel(
    const unsigned short* __restrict__ qw, const unsigned short* __restrict__ kw,
    const unsigned short* __restrict__ vw, float* __restrict__ out)
{
  __shared__ __align__(16) unsigned short kbuf[2][32 * 64];   // [key][hd]
  __shared__ __align__(16) unsigned short vstage[2][32 * 64]; // raw v, [key][hd]
  __shared__ __align__(16) unsigned short vbufT[2][64 * 32];  // [hd][key]
  __shared__ __align__(16) unsigned short pbuf[8][16 * 32];   // per-wave P scratch

  const int tid   = threadIdx.x;
  const int wave  = tid >> 5;
  const int lane  = tid & 31;
  const int l16   = lane & 15;
  const int lhalf = lane >> 4;
  const int tile_base = blockIdx.x * 128;
  const int qbase     = tile_base + wave * 16;
  const size_t brow   = (size_t)blockIdx.y * T_;
  const int srow = tid >> 3, sch = tid & 7;   // staging row/chunk for this thread

  int kofs[8];
#pragma unroll
  for (int j = 0; j < 8; ++j) kofs[j] = 2 * j + ((j >= 4) ? 8 : 0) + lhalf * 8;

  // Q fragments (K = 0..31 and 32..63), loaded once, kept in VGPRs.
  v8u qa0, qa1;
  const unsigned short* qr = qw + (brow + qbase + l16) * HD_;
#pragma unroll
  for (int j = 0; j < 8; ++j) {
    qa0[j] = *(const unsigned int*)(qr + kofs[j]);
    qa1[j] = *(const unsigned int*)(qr + 32 + kofs[j]);
  }
  const v16bf QA0 = __builtin_bit_cast(v16bf, qa0);
  const v16bf QA1 = __builtin_bit_cast(v16bf, qa1);

  v8f acc[4];
  float mrow[8], lrow[8];
#pragma unroll
  for (int t = 0; t < 4; ++t)
#pragma unroll
    for (int r = 0; r < 8; ++r) acc[t][r] = 0.f;
#pragma unroll
  for (int r = 0; r < 8; ++r) { mrow[r] = -3.0e38f; lrow[r] = 0.f; }

  const int nblk = tile_base / 32 + 4;   // keys 0 .. tile_base+127
  const float scale = 0.125f;            // HD^-0.5

  // Prologue: async-stage block 0 into buffer 0.
  {
    const unsigned short* ksrc = kw + (brow + srow) * HD_ + sch * 8;
    const unsigned short* vsrc = vw + (brow + srow) * HD_ + sch * 8;
    ASYNC_CP_B128(&kbuf[0][srow * 64 + sch * 8], ksrc);
    ASYNC_CP_B128(&vstage[0][srow * 64 + sch * 8], vsrc);
  }

  for (int blk = 0; blk < nblk; ++blk) {
    const int cur = blk & 1;
    const int s0  = blk * 32;

    WAIT_ASYNC0();        // this wave's async copies for block `blk` are done
    __syncthreads();      // all waves' copies visible; prior reads of buffers done

    // Overlap: issue async stage of the NEXT block into the other buffer.
    if (blk + 1 < nblk) {
      const unsigned short* ksrc = kw + (brow + s0 + 32 + srow) * HD_ + sch * 8;
      const unsigned short* vsrc = vw + (brow + s0 + 32 + srow) * HD_ + sch * 8;
      ASYNC_CP_B128(&kbuf[cur ^ 1][srow * 64 + sch * 8], ksrc);
      ASYNC_CP_B128(&vstage[cur ^ 1][srow * 64 + sch * 8], vsrc);
    }

    // Transpose this thread's own async-staged v chunk: [key][hd] -> [hd][key].
    {
      const uint4 vv = *(const uint4*)&vstage[cur][srow * 64 + sch * 8];
      const unsigned int* e = (const unsigned int*)&vv;
#pragma unroll
      for (int t = 0; t < 4; ++t) {
        unsigned int w = e[t];
        vbufT[cur][(sch * 8 + 2 * t + 0) * 32 + srow] = (unsigned short)(w & 0xFFFFu);
        vbufT[cur][(sch * 8 + 2 * t + 1) * 32 + srow] = (unsigned short)(w >> 16);
      }
    }
    __syncthreads();      // transposed v visible to all waves

    // S = Q * K^T : two 16x16 tiles (keys 0..15, 16..31), K-dim 64 = 2 WMMAs each.
    v8f sh[2];
#pragma unroll
    for (int h = 0; h < 2; ++h) {
      v8u b0, b1;
      const unsigned short* kr = &kbuf[cur][(h * 16 + l16) * 64];
#pragma unroll
      for (int j = 0; j < 8; ++j) {
        b0[j] = *(const unsigned int*)(kr + kofs[j]);
        b1[j] = *(const unsigned int*)(kr + 32 + kofs[j]);
      }
      v8f z;
#pragma unroll
      for (int r = 0; r < 8; ++r) z[r] = 0.f;
      v8f c0 = __builtin_amdgcn_wmma_f32_16x16x32_bf16(
          false, QA0, false, __builtin_bit_cast(v16bf, b0), (short)0, z, false, false);
      sh[h] = __builtin_amdgcn_wmma_f32_16x16x32_bf16(
          false, QA1, false, __builtin_bit_cast(v16bf, b1), (short)0, c0, false, false);
    }

    // Causal mask + scale. Element (m,n): m = r + 8*lhalf, n = h*16 + l16.
#pragma unroll
    for (int h = 0; h < 2; ++h)
#pragma unroll
      for (int r = 0; r < 8; ++r) {
        int scol = s0 + h * 16 + l16;
        int qrow = qbase + r + 8 * lhalf;
        float v = sh[h][r] * scale;
        sh[h][r] = (scol <= qrow) ? v : -3.0e38f;
      }

    // Row max across the 16 lanes of each half (xor masks 1..8 stay in-half).
    float bm[8];
#pragma unroll
    for (int r = 0; r < 8; ++r) bm[r] = fmaxf(sh[0][r], sh[1][r]);
#pragma unroll
    for (int m = 1; m < 16; m <<= 1)
#pragma unroll
      for (int r = 0; r < 8; ++r) bm[r] = fmaxf(bm[r], __shfl_xor(bm[r], m, 32));

    float sf[8];
#pragma unroll
    for (int r = 0; r < 8; ++r) {
      float mn = fmaxf(mrow[r], bm[r]);
      sf[r]    = __expf(mrow[r] - mn);
      mrow[r]  = mn;
    }

    // P = exp(S - m), row sums, online-l update, rescale O.
#pragma unroll
    for (int h = 0; h < 2; ++h)
#pragma unroll
      for (int r = 0; r < 8; ++r) sh[h][r] = __expf(sh[h][r] - mrow[r]);
    float rs[8];
#pragma unroll
    for (int r = 0; r < 8; ++r) rs[r] = sh[0][r] + sh[1][r];
#pragma unroll
    for (int m = 1; m < 16; m <<= 1)
#pragma unroll
      for (int r = 0; r < 8; ++r) rs[r] += __shfl_xor(rs[r], m, 32);
#pragma unroll
    for (int r = 0; r < 8; ++r) lrow[r] = lrow[r] * sf[r] + rs[r];
#pragma unroll
    for (int t = 0; t < 4; ++t)
#pragma unroll
      for (int r = 0; r < 8; ++r) acc[t][r] *= sf[r];

    // C-layout -> A-layout transpose of P through per-wave LDS scratch (bf16).
    unsigned short* pb = pbuf[wave];
#pragma unroll
    for (int h = 0; h < 2; ++h)
#pragma unroll
      for (int r = 0; r < 8; ++r)
        pb[(r + 8 * lhalf) * 32 + h * 16 + l16] = f2bf(sh[h][r]);
    WAIT_DS0();           // wave-private region: DS in-order, block compiler reorder

    v8u pa;
    const unsigned short* pr = pb + l16 * 32;
#pragma unroll
    for (int j = 0; j < 8; ++j) pa[j] = *(const unsigned int*)(pr + kofs[j]);
    const v16bf PA = __builtin_bit_cast(v16bf, pa);

    // O += P * V, four 16x16 output tiles over HD, K = 32 keys.
#pragma unroll
    for (int t = 0; t < 4; ++t) {
      v8u vb;
      const unsigned short* vr = &vbufT[cur][(t * 16 + l16) * 32];
#pragma unroll
      for (int j = 0; j < 8; ++j) vb[j] = *(const unsigned int*)(vr + kofs[j]);
      acc[t] = __builtin_amdgcn_wmma_f32_16x16x32_bf16(
          false, PA, false, __builtin_bit_cast(v16bf, vb), (short)0, acc[t], false, false);
    }
  }

  float inv[8];
#pragma unroll
  for (int r = 0; r < 8; ++r) inv[r] = 1.0f / lrow[r];
#pragma unroll
  for (int t = 0; t < 4; ++t)
#pragma unroll
    for (int r = 0; r < 8; ++r)
      out[(brow + qbase + r + 8 * lhalf) * HD_ + t * 16 + l16] = acc[t][r] * inv[r];
}

// ---------------------------------------------------------------------------
extern "C" void kernel_launch(void* const* d_in, const int* in_sizes, int n_in,
                              void* d_out, int out_size, void* d_ws, size_t ws_size,
                              hipStream_t stream) {
  const float* x  = (const float*)d_in[0];
  // d_in[1] = freqs_cis (unused by the reference module)
  const float* Wq = (const float*)d_in[2];
  const float* Wk = (const float*)d_in[3];
  const float* Wv = (const float*)d_in[4];
  float* out = (float*)d_out;

  unsigned short* qw = (unsigned short*)d_ws;               // bf16 [BT][64]
  unsigned short* kw = qw + (size_t)B_ * T_ * HD_;
  unsigned short* vw = kw + (size_t)B_ * T_ * HD_;

  qkv_proj_kernel<<<dim3((B_ * T_) / 128, 3), 256, 0, stream>>>(
      x, Wq, Wk, Wv, qw, kw, vw);
  attn_kernel<<<dim3(T_ / 128, B_), 256, 0, stream>>>(qw, kw, vw, out);
}
